// MultiLayerRNN_14242111554075
// MI455X (gfx1250) — matrix operand order repaired and checked by reading, not compiled, not run on hardware
//
#include <hip/hip_runtime.h>
#include <hip/hip_bf16.h>

// Problem dims (fixed by reference)
#define V_  32000
#define E_  256
#define H_  1024
#define B_  16
#define T_  256

typedef __attribute__((ext_vector_type(16))) __bf16 v16bf;
typedef __attribute__((ext_vector_type(8)))  float  v8f;

union Frag { uint4 u[2]; v16bf v; };

// ---- WMMA fragment loaders (CDNA5 16-bit layouts, wave32) ----
// A (16x32, MxK): lane L holds row M=L&15; K = (L>>4)*8 + {0..7} in v0..3,
// and +16 in v4..7  -> two 16-byte contiguous loads (p, p+32B).
__device__ __forceinline__ v16bf load_a_at(const char* p) {
  Frag f;
  f.u[0] = *(const uint4*)(p);
  f.u[1] = *(const uint4*)(p + 32);
  return f.v;
}
// B (32x16, KxN) = W^T with W row-major [rows, K]:
// lane L holds column N=L&15 (weight row r0+N); K = (L>>4)*8 + {0..7} in v0..3,
// and +16 in v4..7 -> two 16-byte loads (p, p+32B).
__device__ __forceinline__ v16bf load_b_at(const char* p) {
  Frag f;
  f.u[0] = *(const uint4*)(p);
  f.u[1] = *(const uint4*)(p + 32);
  return f.v;
}
// Per-lane byte offsets into a row-major [rows, K] bf16 array.
__device__ __forceinline__ const char* a_base(const __hip_bfloat16* base,
                                              int strideK, int lane) {
  int m  = lane & 15;
  int kb = (lane >> 4) * 8;
  return (const char*)(base + (size_t)m * strideK + kb);
}
__device__ __forceinline__ const char* b_base(const __hip_bfloat16* w,
                                              int strideK, int r0, int lane) {
  int r  = r0 + (lane & 15);
  int kb = (lane >> 4) * 8;
  return (const char*)(w + (size_t)r * strideK + kb);
}

__device__ __forceinline__ v8f wmma_bf16(v16bf a, v16bf b, v8f c) {
  return __builtin_amdgcn_wmma_f32_16x16x32_bf16(false, a, false, b,
                                                 (short)0, c, false, false);
}

// ---- prep kernels ----
__global__ void f32_to_bf16_kernel(const float* __restrict__ in,
                                   __hip_bfloat16* __restrict__ out, int n) {
  int i = blockIdx.x * blockDim.x + threadIdx.x;
  if (i < n) out[i] = __float2bfloat16(in[i]);
}

// xs[(t*B+b)*E + e] = bf16(emb[x[b][t]][e])
__global__ void embed_kernel(const int* __restrict__ x,
                             const float* __restrict__ emb,
                             __hip_bfloat16* __restrict__ xs) {
  int i = blockIdx.x * blockDim.x + threadIdx.x;
  if (i >= T_ * B_ * E_) return;
  int e  = i % E_;
  int tb = i / E_;
  int b  = tb % B_;
  int t  = tb / B_;
  int tok = x[b * T_ + t];
  xs[i] = __float2bfloat16(emb[(size_t)tok * E_ + e]);
}

__global__ void init_state_kernel(const float* __restrict__ h0,
                                  const float* __restrict__ c0,
                                  __hip_bfloat16* h_l0, __hip_bfloat16* h_l1,
                                  float* c_l0, float* c_l1,
                                  float* hf_l0, float* hf_l1) {
  int i = blockIdx.x * blockDim.x + threadIdx.x;
  if (i >= B_ * H_) return;
  h_l0[i] = __float2bfloat16(h0[i]);
  h_l1[i] = __float2bfloat16(h0[B_ * H_ + i]);
  c_l0[i] = c0[i];
  c_l1[i] = c0[B_ * H_ + i];
  hf_l0[i] = h0[i];
  hf_l1[i] = h0[B_ * H_ + i];
}

// ---- LSTM step: grid = H/16 blocks, 128 threads (4 waves = 4 gates) ----
// Gate g tile = sigmoid( xt @ wx[g*H+n0 : +16, :]^T + h @ wh[...]^T + bh ).
// K-loop split into x-part and h-part so the steady state is pure
// load_b128 + pointer-increment + v_wmma (no per-iteration selects).
__global__ __launch_bounds__(128)
void lstm_step_kernel(const __hip_bfloat16* __restrict__ xsrc, int Kx,
                      const __hip_bfloat16* __restrict__ wx,
                      const __hip_bfloat16* __restrict__ wh,
                      const float* __restrict__ bh,
                      const __hip_bfloat16* __restrict__ hprev,
                      __hip_bfloat16* __restrict__ hnext,
                      float* __restrict__ cbuf, float* __restrict__ hf32,
                      __hip_bfloat16* __restrict__ outbuf,
                      int t, int ot, int ob) {
  const int lane = threadIdx.x & 31;
  const int wave = threadIdx.x >> 5;          // gate id: 0=f 1=i 2=o 3=chat
  const int n0   = blockIdx.x * 16;           // hidden-unit tile
  const int r0   = wave * H_ + n0;            // weight-row tile for this gate

  const __hip_bfloat16* xt = xsrc + (size_t)t * B_ * Kx;

  v8f acc = {};

  // x contribution (K = Kx)
  {
    const char* pa = a_base(xt, Kx, lane);
    const char* pb = b_base(wx, Kx, r0, lane);
    for (int kt = 0; kt < Kx / 32; ++kt) {
      v16bf a = load_a_at(pa);
      v16bf b = load_b_at(pb);
      acc = wmma_bf16(a, b, acc);
      pa += 64;                                // 32 bf16 per K-step
      pb += 64;
    }
  }
  // recurrent contribution (K = H)
  {
    const char* pa = a_base(hprev, H_, lane);
    const char* pb = b_base(wh, H_, r0, lane);
    #pragma unroll 2
    for (int kt = 0; kt < H_ / 32; ++kt) {
      v16bf a = load_a_at(pa);
      v16bf b = load_b_at(pb);
      acc = wmma_bf16(a, b, acc);
      pa += 64;
      pb += 64;
    }
  }

  const float bias = bh[r0 + (lane & 15)];
  __shared__ float gbuf[4][16][16];           // [gate][batch][hidden]
  #pragma unroll
  for (int r = 0; r < 8; ++r) {
    int m = r + 8 * (lane >> 4);              // C/D layout: VGPR r, lane halves
    float g = acc[r] + bias;
    g = 1.0f / (1.0f + __expf(-g));           // all 4 gates use sigmoid (ref quirk)
    gbuf[wave][m][lane & 15] = g;
  }
  __syncthreads();

  for (int idx = threadIdx.x; idx < 256; idx += 128) {
    int m = idx >> 4, n = idx & 15;
    int col = n0 + n;
    float f  = gbuf[0][m][n];
    float i  = gbuf[1][m][n];
    float o  = gbuf[2][m][n];
    float ch = gbuf[3][m][n];
    float cn = f * cbuf[m * H_ + col] + i * ch;
    float hn = o * tanhf(cn);
    cbuf[m * H_ + col] = cn;
    hf32[m * H_ + col] = hn;
    hnext[m * H_ + col] = __float2bfloat16(hn);
    // layer output is the o-gate (ref quirk)
    outbuf[(size_t)m * ob + (size_t)t * ot + col] = __float2bfloat16(o);
  }
}

// ---- head GEMM: logits[m, v] = acts[m,:] . head_w[v,:] + head_b[v] ----
// acts: [B*T, H] bf16 row-major (m = b*T + t); out: [B,T,V] f32 = [m, V].
// 2x2 register tile per wave: 2 M-tiles x 2 N-tiles -> 4 fragment loads
// feed 4 WMMAs (2x data reuse vs 1x1).
__global__ __launch_bounds__(256)
void head_kernel(const __hip_bfloat16* __restrict__ acts,
                 const __hip_bfloat16* __restrict__ hw,
                 const float* __restrict__ hb,
                 float* __restrict__ out) {
  const int lane = threadIdx.x & 31;
  const int wave = threadIdx.x >> 5;
  const int mt2 = blockIdx.x & 127;           // 128 M-pair tiles (32 rows each)
  const int ng  = blockIdx.x >> 7;            // 125 N groups
  const int nt2 = ng * 8 + wave;              // 1000 N-pair tiles (32 cols each)
  const int m0  = mt2 * 32;
  const int v0  = nt2 * 32;

  v8f acc00 = {}, acc01 = {}, acc10 = {}, acc11 = {};

  const char* pa0 = a_base(acts + (size_t)m0 * H_, H_, lane);
  const char* pa1 = a_base(acts + (size_t)(m0 + 16) * H_, H_, lane);
  const char* pb0 = b_base(hw, H_, v0, lane);
  const char* pb1 = b_base(hw, H_, v0 + 16, lane);

  #pragma unroll 2
  for (int kt = 0; kt < H_ / 32; ++kt) {
    v16bf a0 = load_a_at(pa0);
    v16bf a1 = load_a_at(pa1);
    v16bf b0 = load_b_at(pb0);
    v16bf b1 = load_b_at(pb1);
    acc00 = wmma_bf16(a0, b0, acc00);
    acc01 = wmma_bf16(a0, b1, acc01);
    acc10 = wmma_bf16(a1, b0, acc10);
    acc11 = wmma_bf16(a1, b1, acc11);
    pa0 += 64; pa1 += 64; pb0 += 64; pb1 += 64;
  }

  const int   vc0 = v0 + (lane & 15);
  const int   vc1 = vc0 + 16;
  const float bias0 = hb[vc0];
  const float bias1 = hb[vc1];
  #pragma unroll
  for (int r = 0; r < 8; ++r) {
    int ma = m0 + r + 8 * (lane >> 4);
    int mb = ma + 16;
    out[(size_t)ma * V_ + vc0] = acc00[r] + bias0;
    out[(size_t)ma * V_ + vc1] = acc01[r] + bias1;
    out[(size_t)mb * V_ + vc0] = acc10[r] + bias0;
    out[(size_t)mb * V_ + vc1] = acc11[r] + bias1;
  }
}

__global__ void finalize_state_kernel(const float* hf_l0, const float* hf_l1,
                                      const float* c_l0, const float* c_l1,
                                      float* out_state) {
  int i = blockIdx.x * blockDim.x + threadIdx.x;
  if (i >= B_ * H_) return;
  out_state[i]               = hf_l0[i];
  out_state[B_ * H_ + i]     = hf_l1[i];
  out_state[2 * B_ * H_ + i] = c_l0[i];
  out_state[3 * B_ * H_ + i] = c_l1[i];
}

extern "C" void kernel_launch(void* const* d_in, const int* in_sizes, int n_in,
                              void* d_out, int out_size, void* d_ws, size_t ws_size,
                              hipStream_t stream) {
  (void)in_sizes; (void)n_in; (void)out_size; (void)ws_size;
  const int*   x      = (const int*)  d_in[0];
  const float* h0     = (const float*)d_in[1];
  const float* c0     = (const float*)d_in[2];
  const float* emb    = (const float*)d_in[3];
  const float* wx0    = (const float*)d_in[4];
  const float* wh0    = (const float*)d_in[5];
  const float* bh0    = (const float*)d_in[6];
  const float* wx1    = (const float*)d_in[7];
  const float* wh1    = (const float*)d_in[8];
  const float* bh1    = (const float*)d_in[9];
  const float* head_w = (const float*)d_in[10];
  const float* head_b = (const float*)d_in[11];

  char* ws = (char*)d_ws;
  size_t off = 0;
  auto alloc = [&](size_t bytes) -> char* {
    char* p = ws + off;
    off = (off + bytes + 255) & ~(size_t)255;
    return p;
  };

  __hip_bfloat16* xs    = (__hip_bfloat16*)alloc((size_t)T_ * B_ * E_ * 2);  // [T,B,E]
  __hip_bfloat16* o0buf = (__hip_bfloat16*)alloc((size_t)T_ * B_ * H_ * 2);  // [T,B,H]
  __hip_bfloat16* acts  = (__hip_bfloat16*)alloc((size_t)B_ * T_ * H_ * 2);  // [B*T,H]
  __hip_bfloat16* wx0q  = (__hip_bfloat16*)alloc((size_t)4 * H_ * E_ * 2);
  __hip_bfloat16* wh0q  = (__hip_bfloat16*)alloc((size_t)4 * H_ * H_ * 2);
  __hip_bfloat16* wx1q  = (__hip_bfloat16*)alloc((size_t)4 * H_ * H_ * 2);
  __hip_bfloat16* wh1q  = (__hip_bfloat16*)alloc((size_t)4 * H_ * H_ * 2);
  __hip_bfloat16* hwq   = (__hip_bfloat16*)alloc((size_t)V_ * H_ * 2);
  __hip_bfloat16* h0a   = (__hip_bfloat16*)alloc((size_t)B_ * H_ * 2);
  __hip_bfloat16* h0b   = (__hip_bfloat16*)alloc((size_t)B_ * H_ * 2);
  __hip_bfloat16* h1a   = (__hip_bfloat16*)alloc((size_t)B_ * H_ * 2);
  __hip_bfloat16* h1b   = (__hip_bfloat16*)alloc((size_t)B_ * H_ * 2);
  float* c_l0  = (float*)alloc((size_t)B_ * H_ * 4);
  float* c_l1  = (float*)alloc((size_t)B_ * H_ * 4);
  float* hf_l0 = (float*)alloc((size_t)B_ * H_ * 4);
  float* hf_l1 = (float*)alloc((size_t)B_ * H_ * 4);

  const int cvt_threads = 256;
  auto cvt = [&](const float* in, __hip_bfloat16* out, int n) {
    f32_to_bf16_kernel<<<(n + cvt_threads - 1) / cvt_threads, cvt_threads, 0, stream>>>(in, out, n);
  };
  cvt(wx0, wx0q, 4 * H_ * E_);
  cvt(wh0, wh0q, 4 * H_ * H_);
  cvt(wx1, wx1q, 4 * H_ * H_);
  cvt(wh1, wh1q, 4 * H_ * H_);
  cvt(head_w, hwq, V_ * H_);

  {
    int n = T_ * B_ * E_;
    embed_kernel<<<(n + 255) / 256, 256, 0, stream>>>(x, emb, xs);
  }
  {
    int n = B_ * H_;
    init_state_kernel<<<(n + 255) / 256, 256, 0, stream>>>(h0, c0, h0a, h1a,
                                                           c_l0, c_l1, hf_l0, hf_l1);
  }

  // Layer 0: x from embeddings (Kx=E), output -> o0buf [T,B,H] (ot=B*H, ob=H)
  {
    __hip_bfloat16* hp = h0a; __hip_bfloat16* hn = h0b;
    for (int t = 0; t < T_; ++t) {
      lstm_step_kernel<<<H_ / 16, 128, 0, stream>>>(xs, E_, wx0q, wh0q, bh0,
                                                    hp, hn, c_l0, hf_l0,
                                                    o0buf, t, B_ * H_, H_);
      __hip_bfloat16* tmp = hp; hp = hn; hn = tmp;
    }
  }
  // Layer 1: x from o0buf (Kx=H), output -> acts [B*T,H] (ot=H, ob=T*H)
  {
    __hip_bfloat16* hp = h1a; __hip_bfloat16* hn = h1b;
    for (int t = 0; t < T_; ++t) {
      lstm_step_kernel<<<H_ / 16, 128, 0, stream>>>(o0buf, H_, wx1q, wh1q, bh1,
                                                    hp, hn, c_l1, hf_l1,
                                                    acts, t, H_, T_ * H_);
      __hip_bfloat16* tmp = hp; hp = hn; hn = tmp;
    }
  }

  // Head: 128 M-pair tiles x 1000 N-pair tiles, 8 wave-tiles per block.
  head_kernel<<<128 * 125, 256, 0, stream>>>(acts, hwq, head_b, (float*)d_out);

  // States appended after logits.
  float* out_state = (float*)d_out + (size_t)B_ * T_ * V_;
  {
    int n = B_ * H_;
    finalize_state_kernel<<<(n + 255) / 256, 256, 0, stream>>>(hf_l0, hf_l1,
                                                               c_l0, c_l1, out_state);
  }
}
